// TripletMixerMambaLayer_33560874451305
// MI455X (gfx1250) — compile-verified, matrix operand order checked by reading, not run on hardware
//
#include <hip/hip_runtime.h>
#include <hip/hip_bf16.h>
#include <cstdint>

// ---------------------------------------------------------------------------
// Problem constants (TripletMixerMambaLayer)
// ---------------------------------------------------------------------------
#define BSZ     2
#define CDIM    256
#define LLEN    4096            // 64*64
#define BL      (BSZ * LLEN)    // 8192 tokens
#define DIN     512             // D_INNER
#define DSTATE  16
#define DTRANK  16
#define NXPROJ  48              // DT_RANK + 2*D_STATE

typedef __attribute__((ext_vector_type(16))) _Float16 v16h;
typedef __attribute__((ext_vector_type(8)))  _Float16 v8h;
typedef __attribute__((ext_vector_type(8)))  float    v8f;
typedef __attribute__((ext_vector_type(4)))  unsigned int v4u;
typedef __attribute__((ext_vector_type(8)))  int      v8i;
typedef __attribute__((ext_vector_type(4)))  int      v4i;

// ---------------------------------------------------------------------------
// TDM: 2D tile load (global -> LDS), element size = 2 bytes (f16).
// Descriptor per CDNA5 ISA. global_addr = tile start,
// tensor dims given as "remaining" extents so OOB rows/cols read as zero.
// ---------------------------------------------------------------------------
__device__ __forceinline__ void tdm_load_tile_f16(
    const void* gptr, unsigned lds_off,
    unsigned rem_cols,            // tensor_dim0 (elements)
    unsigned rem_rows,            // tensor_dim1 (rows)
    unsigned long long row_stride_elems,
    unsigned tile_cols, unsigned tile_rows)
{
    unsigned long long ga = (unsigned long long)(uintptr_t)gptr;
    v4u g0;
    g0[0] = 1u;                                        // count=1, is_restore=0
    g0[1] = lds_off;                                   // lds_addr (bytes)
    g0[2] = (unsigned)(ga & 0xFFFFFFFFull);            // global_addr[31:0]
    g0[3] = (unsigned)((ga >> 32) & 0x01FFFFFFull)     // global_addr[56:32]
          | (2u << 30);                                // type = 2 ("image")
    v8i g1;
    g1[0] = (int)(1u << 16);                           // data_size=1 (2B)
    g1[1] = (int)((rem_cols & 0xFFFFu) << 16);         // tensor_dim0[15:0]
    g1[2] = (int)(((rem_cols >> 16) & 0xFFFFu) |
                  ((rem_rows & 0xFFFFu) << 16));       // dim0[31:16] | dim1[15:0]
    g1[3] = (int)(((rem_rows >> 16) & 0xFFFFu) |
                  ((tile_cols & 0xFFFFu) << 16));      // dim1[31:16] | tile_dim0
    g1[4] = (int)(tile_rows & 0xFFFFu);                // tile_dim1 (tile_dim2=0)
    g1[5] = (int)(unsigned)(row_stride_elems & 0xFFFFFFFFull); // dim0_stride[31:0]
    g1[6] = (int)(unsigned)((row_stride_elems >> 32) & 0xFFFFull);
    g1[7] = 0;
    v4i z4 = {0, 0, 0, 0};
#if __clang_major__ >= 23
    v8i z8 = {0, 0, 0, 0, 0, 0, 0, 0};
    __builtin_amdgcn_tensor_load_to_lds(g0, g1, z4, z4, z8, 0);
#else
    __builtin_amdgcn_tensor_load_to_lds(g0, g1, z4, z4, 0);
#endif
}

__device__ __forceinline__ v16h load_frag(const _Float16* p)
{
    union { v16h v; v8h h[2]; } u;
    u.h[0] = *(const v8h*)(p);
    u.h[1] = *(const v8h*)(p + 16);
    return u.v;
}

// ---------------------------------------------------------------------------
// WMMA GEMM:  C[M,N](f32)  (+)=  A[M,K](f16 row major) * Bw[N,K]^T
//
// Block = 256 threads = 8 waves. Block tile 64(M) x 128(N), K-step 64.
// Waves arranged 2(M) x 4(N); each wave owns a 32x32 output (4 acc tiles).
// Per K-step: 8 fragments feed 8 v_wmma_f32_16x16x32_f16 (reuse factor 2),
// one barrier pair per 8 WMMAs. A/B tiles are staged into double-buffered
// LDS by wave 0 via the Tensor Data Mover; the next stage is issued BEFORE
// computing the current buffer so the DMA overlaps the WMMA pipe, and
// s_wait_tensorcnt lands after the compute.
// Requires: K % 64 == 0, M % 64 == 0 (true for every call here).
// ---------------------------------------------------------------------------
#define GEMM_TM 64
#define GEMM_TN 128
#define GEMM_TK 64

__global__ __launch_bounds__(256)
void gemm_f16_wmma(const _Float16* __restrict__ A,
                   const _Float16* __restrict__ Bw,
                   float* __restrict__ Cout,
                   int M, int N, int K, int accum)
{
    __shared__ _Float16 sA[2][GEMM_TM * GEMM_TK];   // 2 x 8 KB
    __shared__ _Float16 sB[2][GEMM_TN * GEMM_TK];   // 2 x 16 KB

    const int tid  = threadIdx.x;
    const int wave = tid >> 5;          // 0..7
    const int lane = tid & 31;
    const int wm   = wave & 1;          // M sub-tile (0..1) -> 32 rows
    const int wn   = wave >> 1;         // N sub-tile (0..3) -> 32 cols
    const int m0   = blockIdx.x * GEMM_TM;
    const int n0   = blockIdx.y * GEMM_TN;

    const int r16  = lane & 15;         // row (A) / col (B) within 16
    const int kh   = lane >> 4;         // K-half selector per ISA A layout

    const unsigned ldsA0 = (unsigned)(unsigned long long)(uintptr_t)&sA[0][0];
    const unsigned ldsA1 = (unsigned)(unsigned long long)(uintptr_t)&sA[1][0];
    const unsigned ldsB0 = (unsigned)(unsigned long long)(uintptr_t)&sB[0][0];
    const unsigned ldsB1 = (unsigned)(unsigned long long)(uintptr_t)&sB[1][0];

    // C/D layout: VGPR r, lanes<16 -> M=r, lanes>=16 -> M=r+8; N = lane&15.
    const int obase_m = m0 + 32 * wm + 8 * kh;
    const int obase_n = n0 + 32 * wn + r16;

    v8f acc[2][2];
    #pragma unroll
    for (int mi = 0; mi < 2; ++mi)
        #pragma unroll
        for (int ni = 0; ni < 2; ++ni) {
            int ocol = obase_n + 16 * ni;
            if (accum && ocol < N) {
                #pragma unroll
                for (int r = 0; r < 8; ++r)
                    acc[mi][ni][r] =
                        Cout[(size_t)(obase_m + 16 * mi + r) * N + ocol];
            } else {
                #pragma unroll
                for (int r = 0; r < 8; ++r) acc[mi][ni][r] = 0.0f;
            }
        }

    // Prologue: stage K-chunk 0 into buffer 0.
    if (wave == 0) {
        tdm_load_tile_f16(A + (size_t)m0 * K, ldsA0,
                          (unsigned)K, (unsigned)(M - m0),
                          (unsigned long long)K, GEMM_TK, GEMM_TM);
        int remN = N - n0; if (remN < 1) remN = 1;
        tdm_load_tile_f16(Bw + (size_t)n0 * K, ldsB0,
                          (unsigned)K, (unsigned)remN,
                          (unsigned long long)K, GEMM_TK, GEMM_TN);
        __builtin_amdgcn_s_wait_tensorcnt(0);
    }
    __syncthreads();

    int cur = 0;
    for (int k0 = 0; k0 < K; k0 += GEMM_TK) {
        const int nxt = cur ^ 1;
        const bool more = (k0 + GEMM_TK) < K;

        // Issue next stage ASYNC before computing the current buffer.
        if (wave == 0 && more) {
            tdm_load_tile_f16(A + (size_t)m0 * K + (k0 + GEMM_TK),
                              nxt ? ldsA1 : ldsA0,
                              (unsigned)(K - k0 - GEMM_TK), (unsigned)(M - m0),
                              (unsigned long long)K, GEMM_TK, GEMM_TM);
            int remN = N - n0; if (remN < 1) remN = 1;
            tdm_load_tile_f16(Bw + (size_t)n0 * K + (k0 + GEMM_TK),
                              nxt ? ldsB1 : ldsB0,
                              (unsigned)(K - k0 - GEMM_TK), (unsigned)remN,
                              (unsigned long long)K, GEMM_TK, GEMM_TN);
        }

        // Compute on the current buffer: 8 WMMAs per K-step.
        const _Float16* cA = &sA[cur][0];
        const _Float16* cB = &sB[cur][0];
        #pragma unroll
        for (int kk = 0; kk < GEMM_TK; kk += 32) {
            v16h af[2], bf[2];
            #pragma unroll
            for (int mi = 0; mi < 2; ++mi)
                af[mi] = load_frag(cA + (32 * wm + 16 * mi + r16) * GEMM_TK
                                      + kk + kh * 8);
            #pragma unroll
            for (int ni = 0; ni < 2; ++ni)
                bf[ni] = load_frag(cB + (32 * wn + 16 * ni + r16) * GEMM_TK
                                      + kk + kh * 8);
            #pragma unroll
            for (int mi = 0; mi < 2; ++mi)
                #pragma unroll
                for (int ni = 0; ni < 2; ++ni)
                    acc[mi][ni] = __builtin_amdgcn_wmma_f32_16x16x32_f16(
                        false, af[mi], false, bf[ni], (short)0,
                        acc[mi][ni], false, false);
        }

        __syncthreads();                     // everyone done reading `cur`
        if (wave == 0 && more)
            __builtin_amdgcn_s_wait_tensorcnt(0);   // `nxt` has landed
        __syncthreads();                     // `nxt` visible to all waves
        cur = nxt;
    }

    #pragma unroll
    for (int mi = 0; mi < 2; ++mi)
        #pragma unroll
        for (int ni = 0; ni < 2; ++ni) {
            int ocol = obase_n + 16 * ni;
            if (ocol < N) {
                #pragma unroll
                for (int r = 0; r < 8; ++r)
                    Cout[(size_t)(obase_m + 16 * mi + r) * N + ocol] =
                        acc[mi][ni][r];
            }
        }
}

// ---------------------------------------------------------------------------
// f32 -> f16 convert (weights)
// ---------------------------------------------------------------------------
__global__ void cvt_f16_kernel(const float* __restrict__ src,
                               _Float16* __restrict__ dst, int n)
{
    int i = blockIdx.x * 256 + threadIdx.x;
    if (i < n) dst[i] = (_Float16)src[i];
}

// ---------------------------------------------------------------------------
// LayerNorm over channels: x(b,c,l) -> xn(token,c) f32 + f16
// One thread per token; channel loop is coalesced over l across the wave.
// ---------------------------------------------------------------------------
__global__ void layernorm_kernel(const float* __restrict__ x,
                                 const float* __restrict__ lnw,
                                 const float* __restrict__ lnb,
                                 float* __restrict__ xn,
                                 _Float16* __restrict__ xnh)
{
    int t = blockIdx.x * 256 + threadIdx.x;
    if (t >= BL) return;
    int b = t >> 12, l = t & (LLEN - 1);
    const float* xp = x + (size_t)b * CDIM * LLEN + l;
    float s = 0.f, s2 = 0.f;
    for (int c = 0; c < CDIM; ++c) {
        float v = xp[(size_t)c * LLEN];
        s += v; s2 += v * v;
    }
    float mu  = s * (1.0f / CDIM);
    float var = s2 * (1.0f / CDIM) - mu * mu;
    float inv = rsqrtf(var + 1e-5f);
    float*    op = xn  + (size_t)t * CDIM;
    _Float16* oh = xnh + (size_t)t * CDIM;
    for (int c = 0; c < CDIM; ++c) {
        float v = (xp[(size_t)c * LLEN] - mu) * inv * lnw[c] + lnb[c];
        op[c] = v; oh[c] = (_Float16)v;
    }
}

// ---------------------------------------------------------------------------
// Causal depthwise conv (k=4, pad left 3) + SiLU on xi = xz[:, :512]
// ---------------------------------------------------------------------------
__global__ void causal_conv_silu_kernel(const float* __restrict__ xz,
                                        const float* __restrict__ w,
                                        const float* __restrict__ bias,
                                        float* __restrict__ xc,
                                        _Float16* __restrict__ xch)
{
    int idx = blockIdx.x * 256 + threadIdx.x;   // t*512 + d
    if (idx >= BL * DIN) return;
    int t = idx >> 9, d = idx & (DIN - 1);
    int l = t & (LLEN - 1);
    float a = bias[d];
    #pragma unroll
    for (int j = 0; j < 4; ++j) {
        int lj = l - 3 + j;
        if (lj >= 0)
            a += w[d * 4 + j] * xz[(size_t)(t - 3 + j) * (2 * DIN) + d];
    }
    float s = a / (1.0f + __expf(-a));
    xc[idx]  = s;
    xch[idx] = (_Float16)s;
}

// ---------------------------------------------------------------------------
// delta = softplus(dt @ W_dt^T + b_dt) : K = 16, scalar FMAs
// ---------------------------------------------------------------------------
__global__ void delta_kernel(const float* __restrict__ xdbl,
                             const float* __restrict__ Wdt,
                             const float* __restrict__ bdt,
                             float* __restrict__ delta)
{
    int idx = blockIdx.x * 256 + threadIdx.x;   // t*512 + d
    if (idx >= BL * DIN) return;
    int t = idx >> 9, d = idx & (DIN - 1);
    const float* xr = xdbl + (size_t)t * NXPROJ;
    float a = bdt[d];
    #pragma unroll
    for (int r = 0; r < DTRANK; ++r)
        a += xr[r] * Wdt[d * DTRANK + r];
    delta[idx] = (a > 20.0f) ? a : log1pf(__expf(a));
}

// ---------------------------------------------------------------------------
// Selective scan: one wave per (b, d); lanes 0..15 hold the 16 SSM states.
// Fuses +D*u and the SiLU(z) gate; emits y in f16 for the W_outp WMMA GEMM.
// ---------------------------------------------------------------------------
__global__ __launch_bounds__(128)
void scan_kernel(const float* __restrict__ delta,
                 const float* __restrict__ xdbl,
                 const float* __restrict__ xc,
                 const float* __restrict__ xz,
                 const float* __restrict__ A_log,
                 const float* __restrict__ D_ssm,
                 _Float16* __restrict__ yh)
{
    int wid = blockIdx.x * 4 + (threadIdx.x >> 5);   // 0..1023
    if (wid >= BSZ * DIN) return;
    int b = wid >> 9, d = wid & (DIN - 1);
    int lane = threadIdx.x & 31;
    int n = lane & 15;

    float a  = -__expf(A_log[d * DSTATE + n]);
    float Dd = D_ssm[d];
    float h  = 0.0f;
    size_t tb = (size_t)b * LLEN;

    for (int l = 0; l < LLEN; ++l) {
        size_t t = tb + l;
        float dt = delta[t * DIN + d];
        float u  = xc[t * DIN + d];
        float Bn = xdbl[t * NXPROJ + DTRANK + n];
        float Cn = xdbl[t * NXPROJ + DTRANK + DSTATE + n];
        h = __expf(dt * a) * h + dt * u * Bn;
        float p = h * Cn;
        p += __shfl_xor(p, 1, 32);
        p += __shfl_xor(p, 2, 32);
        p += __shfl_xor(p, 4, 32);
        p += __shfl_xor(p, 8, 32);
        if (lane == 0) {
            float y = p + Dd * u;
            float z = xz[t * (2 * DIN) + DIN + d];
            y *= z / (1.0f + __expf(-z));
            yh[t * DIN + d] = (_Float16)y;
        }
    }
}

// ---------------------------------------------------------------------------
// Three centered depthwise convs (k=3,5,7) + SiLU, added into combined
// (which already holds x_ssm); also emits f16 copy for the W_out GEMM.
// ---------------------------------------------------------------------------
__global__ void msconv_add_kernel(const float* __restrict__ xn,
                                  const float* __restrict__ c1w, const float* __restrict__ c1b,
                                  const float* __restrict__ c2w, const float* __restrict__ c2b,
                                  const float* __restrict__ c3w, const float* __restrict__ c3b,
                                  float* __restrict__ comb,
                                  _Float16* __restrict__ combh)
{
    int idx = blockIdx.x * 256 + threadIdx.x;   // t*256 + c
    if (idx >= BL * CDIM) return;
    int t = idx >> 8, c = idx & (CDIM - 1);
    int l = t & (LLEN - 1);
    size_t base = (size_t)(t - l) * CDIM;       // batch start token * C
    float a1 = c1b[c], a2 = c2b[c], a3 = c3b[c];
    #pragma unroll
    for (int j = -3; j <= 3; ++j) {
        int lj = l + j;
        if (lj < 0 || lj >= LLEN) continue;
        float v = xn[base + (size_t)lj * CDIM + c];
        if (j >= -1 && j <= 1) a1 += c1w[c * 3 + (j + 1)] * v;
        if (j >= -2 && j <= 2) a2 += c2w[c * 5 + (j + 2)] * v;
        a3 += c3w[c * 7 + (j + 3)] * v;
    }
    float s = a1 / (1.0f + __expf(-a1))
            + a2 / (1.0f + __expf(-a2))
            + a3 / (1.0f + __expf(-a3));
    float v = comb[idx] + s;
    comb[idx]  = v;
    combh[idx] = (_Float16)v;
}

// ---------------------------------------------------------------------------
// out(b,c,l) = t1(token,c) + b_out[c] + b_skip[c] + x(b,c,l)
// ---------------------------------------------------------------------------
__global__ void final_kernel(const float* __restrict__ t1,
                             const float* __restrict__ bout,
                             const float* __restrict__ bskip,
                             const float* __restrict__ x,
                             float* __restrict__ out)
{
    int idx = blockIdx.x * 256 + threadIdx.x;   // (b*256+c)*4096 + l
    if (idx >= BSZ * CDIM * LLEN) return;
    int l = idx & (LLEN - 1);
    int bc = idx >> 12;
    int c = bc & (CDIM - 1);
    int b = bc >> 8;
    size_t tok = ((size_t)b * LLEN + l) * CDIM + c;
    out[idx] = t1[tok] + bout[c] + bskip[c] + x[idx];
}

// ---------------------------------------------------------------------------
// Host launcher
// ---------------------------------------------------------------------------
extern "C" void kernel_launch(void* const* d_in, const int* in_sizes, int n_in,
                              void* d_out, int out_size, void* d_ws, size_t ws_size,
                              hipStream_t stream)
{
    const float* x      = (const float*)d_in[0];
    const float* ln_w   = (const float*)d_in[1];
    const float* ln_b   = (const float*)d_in[2];
    const float* W_in   = (const float*)d_in[3];
    const float* conv_w = (const float*)d_in[4];
    const float* conv_b = (const float*)d_in[5];
    const float* W_xprj = (const float*)d_in[6];
    const float* W_dt   = (const float*)d_in[7];
    const float* b_dt   = (const float*)d_in[8];
    const float* A_log  = (const float*)d_in[9];
    const float* D_ssm  = (const float*)d_in[10];
    const float* W_outp = (const float*)d_in[11];
    const float* c1_w   = (const float*)d_in[12];
    const float* c1_b   = (const float*)d_in[13];
    const float* c2_w   = (const float*)d_in[14];
    const float* c2_b   = (const float*)d_in[15];
    const float* c3_w   = (const float*)d_in[16];
    const float* c3_b   = (const float*)d_in[17];
    const float* W_out  = (const float*)d_in[18];
    const float* b_out  = (const float*)d_in[19];
    const float* W_skip = (const float*)d_in[20];
    const float* b_skip = (const float*)d_in[21];
    float* out = (float*)d_out;

    char* w = (char*)d_ws;
    float*    xn    = (float*)   (w + 0);           //  8 MB
    _Float16* xnh   = (_Float16*)(w + 8388608);     //  4 MB
    float*    xz    = (float*)   (w + 12582912);    // 32 MB
    float*    xc    = (float*)   (w + 46137344);    // 16 MB
    _Float16* xch   = (_Float16*)(w + 62914560);    //  8 MB
    float*    xdbl  = (float*)   (w + 71303168);    // 1.5 MB
    float*    delta = (float*)   (w + 72876032);    // 16 MB
    _Float16* yh    = (_Float16*)(w + 89653248);    //  8 MB
    float*    comb  = (float*)   (w + 98041856);    //  8 MB
    _Float16* combh = (_Float16*)(w + 106430464);   //  4 MB
    float*    t1    = (float*)   (w + 110624768);   //  8 MB
    _Float16* Winh  = (_Float16*)(w + 119013376);
    _Float16* Wxh   = (_Float16*)(w + 119537664);
    _Float16* Woph  = (_Float16*)(w + 119586816);
    _Float16* Wouth = (_Float16*)(w + 119848960);
    _Float16* Wskh  = (_Float16*)(w + 119980032);

    // Weight converts (f32 -> f16)
    cvt_f16_kernel<<<(2*DIN*CDIM + 255)/256, 256, 0, stream>>>(W_in,   Winh,  2*DIN*CDIM);
    cvt_f16_kernel<<<(NXPROJ*DIN  + 255)/256, 256, 0, stream>>>(W_xprj, Wxh,   NXPROJ*DIN);
    cvt_f16_kernel<<<(CDIM*DIN    + 255)/256, 256, 0, stream>>>(W_outp, Woph,  CDIM*DIN);
    cvt_f16_kernel<<<(CDIM*CDIM   + 255)/256, 256, 0, stream>>>(W_out,  Wouth, CDIM*CDIM);
    cvt_f16_kernel<<<(CDIM*CDIM   + 255)/256, 256, 0, stream>>>(W_skip, Wskh,  CDIM*CDIM);

    // LayerNorm
    layernorm_kernel<<<BL/256, 256, 0, stream>>>(x, ln_w, ln_b, xn, xnh);

    // xz = xn @ W_in^T           (M=8192, N=1024, K=256)
    gemm_f16_wmma<<<dim3(BL/GEMM_TM, (2*DIN + GEMM_TN - 1)/GEMM_TN), 256, 0, stream>>>(
        xnh, Winh, xz, BL, 2*DIN, CDIM, 0);

    // causal dwconv + SiLU
    causal_conv_silu_kernel<<<(BL*DIN)/256, 256, 0, stream>>>(xz, conv_w, conv_b, xc, xch);

    // x_dbl = xc @ W_xproj^T     (M=8192, N=48, K=512)
    gemm_f16_wmma<<<dim3(BL/GEMM_TM, (NXPROJ + GEMM_TN - 1)/GEMM_TN), 256, 0, stream>>>(
        xch, Wxh, xdbl, BL, NXPROJ, DIN, 0);

    // delta = softplus(dt @ W_dt^T + b_dt)
    delta_kernel<<<(BL*DIN)/256, 256, 0, stream>>>(xdbl, W_dt, b_dt, delta);

    // selective scan (+ D*u, * SiLU(z)), emit y in f16
    scan_kernel<<<(BSZ*DIN)/4, 128, 0, stream>>>(delta, xdbl, xc, xz, A_log, D_ssm, yh);

    // x_ssm = y @ W_outp^T       (M=8192, N=256, K=512) -> comb
    gemm_f16_wmma<<<dim3(BL/GEMM_TM, (CDIM + GEMM_TN - 1)/GEMM_TN), 256, 0, stream>>>(
        yh, Woph, comb, BL, CDIM, DIN, 0);

    // comb += silu(conv3) + silu(conv5) + silu(conv7)
    msconv_add_kernel<<<(BL*CDIM)/256, 256, 0, stream>>>(
        xn, c1_w, c1_b, c2_w, c2_b, c3_w, c3_b, comb, combh);

    // t1 = comb @ W_out^T        (M=8192, N=256, K=256)
    gemm_f16_wmma<<<dim3(BL/GEMM_TM, (CDIM + GEMM_TN - 1)/GEMM_TN), 256, 0, stream>>>(
        combh, Wouth, t1, BL, CDIM, CDIM, 0);

    // t1 += xn @ W_skip^T        (accumulate)
    gemm_f16_wmma<<<dim3(BL/GEMM_TM, (CDIM + GEMM_TN - 1)/GEMM_TN), 256, 0, stream>>>(
        xnh, Wskh, t1, BL, CDIM, CDIM, 1);

    // out = t1 + b_out + b_skip + x   (transpose back to NCHW)
    final_kernel<<<(BSZ*CDIM*LLEN)/256, 256, 0, stream>>>(t1, b_out, b_skip, x, out);
}
